// TriangleAttentionStartingNode_23708219474364
// MI455X (gfx1250) — matrix operand order repaired and checked by reading, not compiled, not run on hardware
//
#include <hip/hip_runtime.h>

typedef _Float16 f16;
typedef _Float16 v16h __attribute__((ext_vector_type(16)));
typedef _Float16 v8h  __attribute__((ext_vector_type(8)));
typedef float    v8f  __attribute__((ext_vector_type(8)));

#define N_TOK 384
#define DPAIR 128
#define NHEAD 4
#define CHD   32

// ---- WMMA fragment helpers (layouts per CDNA5 ISA 7.12.2) ----

// A 16x32 f16, k order used here (B matches): lane m = lane&15; halves 0..15 -> K = (lane>=16?16:0)+0..15.
// Caller passes rowptr already offset by (lane>=16?16:0); we read rowptr[0..15].
static __device__ __forceinline__ v16h loadA_frag(const f16* rowptr) {
  v8h lo = *(const v8h*)(rowptr);
  v8h hi = *(const v8h*)(rowptr + 8);
  v16h a;
#pragma unroll
  for (int q = 0; q < 8; ++q) { a[q] = lo[q]; a[q + 8] = hi[q]; }
  return a;
}

static __device__ __forceinline__ v8f wmma_f16(v16h a, v16h b, v8f c) {
  return __builtin_amdgcn_wmma_f32_16x16x32_f16(false, a, false, b, (short)0, c,
                                                false, false);
}

static __device__ __forceinline__ float fast_sigmoid(float x) {
  // v_exp_f32 + v_rcp_f32 (avoid the IEEE v_div_scale chain)
  return __builtin_amdgcn_rcpf(1.f + __expf(-x));
}

// ---- Stage 0: transpose weights to col-major f16 (B-fragment friendly) ----
// Wt[512][128]: rows 0..127 Wq-cols (scaled by 1/sqrt(c)), 128..255 Wk,
// 256..383 Wv, 384..511 Wg.  Wot[128][128] = Wo^T.
__global__ void prep_weights(const float* __restrict__ Wq, const float* __restrict__ Wk,
                             const float* __restrict__ Wv, const float* __restrict__ Wg,
                             const float* __restrict__ Wo, f16* __restrict__ Wt,
                             f16* __restrict__ Wot, float qscale) {
  int idx = blockIdx.x * 256 + threadIdx.x;
  if (idx < 512 * 128) {
    int row = idx >> 7, k = idx & 127;
    int sel = row >> 7, col = row & 127;
    const float* W = (sel == 0) ? Wq : (sel == 1) ? Wk : (sel == 2) ? Wv : Wg;
    float v = W[k * 128 + col];
    if (sel == 0) v *= qscale;
    Wt[idx] = (f16)v;
  } else if (idx < 512 * 128 + 128 * 128) {
    int id2 = idx - 512 * 128;
    int n = id2 >> 7, k = id2 & 127;
    Wot[id2] = (f16)Wo[k * 128 + n];
  }
}

// ---- Stage 1: LayerNorm + pair bias + QKVG projection ----
// 64 threads (2 waves), 64 positions per WG (one block never crosses an i row:
// 384 = 6*64).  wave0: Q|K columns, wave1: V|G columns.
// A fragments are loaded from LDS right next to each WMMA chain (short live
// ranges -> no scratch spills); the compiler hoists the 4 B fragments.
__global__ __launch_bounds__(64) void proj_kernel(
    const float* __restrict__ Zraw, const float* __restrict__ gamma,
    const float* __restrict__ beta, const float* __restrict__ bw,
    const float* __restrict__ bg, const f16* __restrict__ Wt,
    f16* __restrict__ Qw, f16* __restrict__ Kw, f16* __restrict__ Vt,
    f16* __restrict__ Gw, float* __restrict__ Bias) {
  __shared__ __align__(16) float Zs[64][132];   // staged f32 rows (pad: banks)
  __shared__ __align__(16) f16 Zln[64][128];    // normalized f16 (A operand)

  const int t = threadIdx.x;
  const int i = blockIdx.x / 6;               // scalar
  const int jbase = (blockIdx.x % 6) * 64;    // scalar

  // Coalesced staging of the 64x128 f32 block (each wave reads 512B runs).
  const float4* src = (const float4*)(Zraw + ((size_t)i * N_TOK + jbase) * DPAIR);
#pragma unroll
  for (int e4 = t, it = 0; it < 32; e4 += 64, ++it) {
    float4 v = src[e4];
    int r = e4 >> 5, c = (e4 & 31) * 4;
    Zs[r][c + 0] = v.x; Zs[r][c + 1] = v.y; Zs[r][c + 2] = v.z; Zs[r][c + 3] = v.w;
  }
  __syncthreads();

  // LayerNorm statistics (f32) from LDS
  float s = 0.f, ss = 0.f;
#pragma unroll 8
  for (int c = 0; c < DPAIR; ++c) {
    float v = Zs[t][c];
    s += v; ss += v * v;
  }
  const float mu = s * (1.f / DPAIR);
  const float var = ss * (1.f / DPAIR) - mu * mu;
  const float rstd = rsqrtf(var + 1e-5f);

  float b0 = 0.f, b1 = 0.f, b2 = 0.f, b3 = 0.f;
#pragma unroll 4
  for (int c = 0; c < DPAIR; ++c) {
    float zn = (Zs[t][c] - mu) * rstd * gamma[c] + beta[c];
    Zln[t][c] = (f16)zn;
    float4 w = *(const float4*)(bw + c * 4);
    b0 += zn * w.x; b1 += zn * w.y; b2 += zn * w.z; b3 += zn * w.w;
  }
  {
    const int j = jbase + t;
    Bias[((size_t)0 * N_TOK + i) * N_TOK + j] = b0;
    Bias[((size_t)1 * N_TOK + i) * N_TOK + j] = b1;
    Bias[((size_t)2 * N_TOK + i) * N_TOK + j] = b2;
    Bias[((size_t)3 * N_TOK + i) * N_TOK + j] = b3;
  }
  __syncthreads();

  const int wave = t >> 5, lane = t & 31;
  const int n15 = lane & 15;
  const int hi8 = (lane >> 4) * 8;   // accumulator row offset (D row = r + hi8)
  const int b16 = (lane >> 4) * 16;  // B fragment K sub-offset

#pragma unroll 1
  for (int nt = 0; nt < 16; ++nt) {
    const int ncol = wave * 256 + nt * 16;
    const int matU = __builtin_amdgcn_readfirstlane(ncol >> 7);  // 0=Q 1=K 2=V 3=G
    const int cw = ncol & 127;
    const int col = cw + n15;
    v16h Bf[4];
#pragma unroll
    for (int kc = 0; kc < 4; ++kc)
      Bf[kc] = *(const v16h*)(Wt + (size_t)(ncol + n15) * 128 + kc * 32 + b16);
#pragma unroll 1
    for (int mt = 0; mt < 4; ++mt) {
      v8f acc = {};
#pragma unroll
      for (int kc = 0; kc < 4; ++kc) {
        v16h A = loadA_frag(&Zln[mt * 16 + n15][kc * 32 + b16]);
        acc = wmma_f16(A, Bf[kc], acc);
      }
      // scalar branch on wave-uniform matrix selector
      if (matU == 0) {
        f16* base = Qw + (((size_t)i * NHEAD + (col >> 5)) * N_TOK) * CHD + (col & 31);
#pragma unroll
        for (int r = 0; r < 8; ++r)
          base[(size_t)(jbase + mt * 16 + r + hi8) * CHD] = (f16)acc[r];
      } else if (matU == 1) {
        f16* base = Kw + (((size_t)i * NHEAD + (col >> 5)) * N_TOK) * CHD + (col & 31);
#pragma unroll
        for (int r = 0; r < 8; ++r)
          base[(size_t)(jbase + mt * 16 + r + hi8) * CHD] = (f16)acc[r];
      } else if (matU == 2) {
        f16* base = Vt + (((size_t)i * NHEAD + (col >> 5)) * CHD + (col & 31)) * N_TOK + jbase;
#pragma unroll
        for (int r = 0; r < 8; ++r)
          base[mt * 16 + r + hi8] = (f16)acc[r];
      } else {
        const float bgc = bg[col];
        f16* base = Gw + ((size_t)i * N_TOK + jbase) * DPAIR + col;
#pragma unroll
        for (int r = 0; r < 8; ++r)
          base[(size_t)(mt * 16 + r + hi8) * DPAIR] = (f16)fast_sigmoid(acc[r] + bgc);
      }
    }
  }
}

// ---- Stage 2: attention + gate + output projection + residual ----
// 128 threads = 4 waves, wave = head; 16 q-rows per WG; grid (24, 384).
__global__ __launch_bounds__(128) void attn_kernel(
    const float* __restrict__ Zraw, const f16* __restrict__ Qw,
    const f16* __restrict__ Kw, const f16* __restrict__ Vt,
    const f16* __restrict__ Gw, const float* __restrict__ Bias,
    const f16* __restrict__ Wot, const float* __restrict__ obias,
    float* __restrict__ out) {
  __shared__ __align__(16) f16 at[NHEAD][16][392];  // scores -> exp weights (f16)
  __shared__ float rinv[NHEAD][16];                 // 1/rowsum
  __shared__ __align__(16) f16 Ost[16][128];        // gated O, all heads

  const int i = blockIdx.y;
  const int q0 = blockIdx.x * 16;
  const int t = threadIdx.x;
  const int h = t >> 5, lane = t & 31;
  const int n15 = lane & 15;
  const int hi8 = (lane >> 4) * 8;
  const int b16 = (lane >> 4) * 16;

  // Q fragment (scale pre-folded into Wq); K=c=32 -> single chunk, keep in regs
  const f16* qrow = Qw + (((size_t)i * NHEAD + h) * N_TOK + q0 + n15) * CHD + b16;
  const v16h Aq = loadA_frag(qrow);
  const f16* krow = Kw + (((size_t)i * NHEAD + h) * N_TOK) * CHD;
  const f16* vrow = Vt + (((size_t)i * NHEAD + h) * CHD) * N_TOK;

  // ---- S = Q K^T + bias (24 tiles of 16 k-positions) ----
#pragma unroll 1
  for (int jt = 0; jt < 24; ++jt) {
    const int j0 = jt * 16;
    v8f c;
    const float* bp = Bias + ((size_t)h * N_TOK + q0) * N_TOK + j0 + n15;
#pragma unroll
    for (int r = 0; r < 8; ++r) c[r] = bp[(size_t)(r + hi8) * N_TOK];
    v16h B = *(const v16h*)(krow + (size_t)(j0 + n15) * CHD + b16);
    c = wmma_f16(Aq, B, c);
#pragma unroll
    for (int r = 0; r < 8; ++r) at[h][r + hi8][j0 + n15] = (f16)c[r];
    if (jt < 23)  // hint next K tile toward the caches (global_prefetch_b8)
      __builtin_prefetch(krow + (size_t)(j0 + 16 + n15) * CHD, 0, 1);
  }
  __builtin_prefetch(vrow + (size_t)n15 * N_TOK + b16 * 2, 0, 1);
  __syncthreads();

  // ---- row softmax: all 32 lanes (half-row each), v8h-vectorized LDS ----
  {
    const int row = lane & 15;
    const int c0 = (lane >> 4) * 192;
    float mx = -3.4e38f;
#pragma unroll 2
    for (int jj = 0; jj < 192; jj += 8) {
      v8h v = *(const v8h*)&at[h][row][c0 + jj];
#pragma unroll
      for (int q = 0; q < 8; ++q) mx = fmaxf(mx, (float)v[q]);
    }
    mx = fmaxf(mx, __shfl_xor(mx, 16, 32));
    float sum = 0.f;
#pragma unroll 2
    for (int jj = 0; jj < 192; jj += 8) {
      v8h v = *(const v8h*)&at[h][row][c0 + jj];
      v8h e;
#pragma unroll
      for (int q = 0; q < 8; ++q) {
        float ev = __expf((float)v[q] - mx);
        sum += ev;
        e[q] = (f16)ev;
      }
      *(v8h*)&at[h][row][c0 + jj] = e;
    }
    sum += __shfl_xor(sum, 16, 32);
    rinv[h][row] = __builtin_amdgcn_rcpf(sum);  // both halves write same value
  }
  __syncthreads();

  // ---- O = P V: dual accumulators, each A fragment feeds both c-tiles ----
  {
    v8f acc0 = {}, acc1 = {};
    const f16* v0p = vrow + (size_t)n15 * N_TOK + b16;
    const f16* v1p = vrow + (size_t)(16 + n15) * N_TOK + b16;
#pragma unroll 1
    for (int kc = 0; kc < 12; ++kc) {
      v16h A = loadA_frag(&at[h][n15][kc * 32 + b16]);
      v16h B0 = *(const v16h*)(v0p + kc * 32);
      acc0 = wmma_f16(A, B0, acc0);
      v16h B1 = *(const v16h*)(v1p + kc * 32);
      acc1 = wmma_f16(A, B1, acc1);
    }
    const f16* gbase = Gw + ((size_t)i * N_TOK + q0) * DPAIR + h * CHD + n15;
#pragma unroll
    for (int r = 0; r < 8; ++r) {
      int m = r + hi8;
      float ri = rinv[h][m];
      float g0 = (float)gbase[(size_t)m * DPAIR];
      float g1 = (float)gbase[(size_t)m * DPAIR + 16];
      Ost[m][h * CHD + n15] = (f16)(acc0[r] * ri * g0);
      Ost[m][h * CHD + 16 + n15] = (f16)(acc1[r] * ri * g1);
    }
  }
  __syncthreads();

  // ---- out = (gate*O) @ Wo + out_bias + Z_raw: dual accumulators ----
  {
    const int ncol0 = (h * 2) * 16;
    const int ncol1 = ncol0 + 16;
    v8f acc0 = {}, acc1 = {};
#pragma unroll
    for (int kc = 0; kc < 4; ++kc) {
      v16h A = loadA_frag(&Ost[n15][kc * 32 + b16]);
      v16h B0 = *(const v16h*)(Wot + (size_t)(ncol0 + n15) * 128 + kc * 32 + b16);
      acc0 = wmma_f16(A, B0, acc0);
      v16h B1 = *(const v16h*)(Wot + (size_t)(ncol1 + n15) * 128 + kc * 32 + b16);
      acc1 = wmma_f16(A, B1, acc1);
    }
    const float ob0 = obias[ncol0 + n15];
    const float ob1 = obias[ncol1 + n15];
#pragma unroll
    for (int r = 0; r < 8; ++r) {
      size_t row = ((size_t)i * N_TOK + q0 + r + hi8) * DPAIR;
      size_t i0 = row + ncol0 + n15;
      size_t i1 = row + ncol1 + n15;
      out[i0] = Zraw[i0] + acc0[r] + ob0;
      out[i1] = Zraw[i1] + acc1[r] + ob1;
    }
  }
}

// ---- host launch ----
extern "C" void kernel_launch(void* const* d_in, const int* in_sizes, int n_in,
                              void* d_out, int out_size, void* d_ws, size_t ws_size,
                              hipStream_t stream) {
  const float* Zraw  = (const float*)d_in[0];
  const float* gamma = (const float*)d_in[1];
  const float* beta  = (const float*)d_in[2];
  const float* bw    = (const float*)d_in[3];
  const float* Wq    = (const float*)d_in[4];
  const float* Wk    = (const float*)d_in[5];
  const float* Wv    = (const float*)d_in[6];
  const float* Wg    = (const float*)d_in[7];
  const float* bg    = (const float*)d_in[8];
  const float* Wo    = (const float*)d_in[9];
  const float* ob    = (const float*)d_in[10];
  float* out = (float*)d_out;
  char* ws = (char*)d_ws;

  const size_t QKVG = (size_t)N_TOK * N_TOK * DPAIR * sizeof(f16);  // 37,748,736
  f16*   Wt  = (f16*)(ws);                         // 512*128*2 = 131072
  f16*   Wot = (f16*)(ws + 131072);                // 128*128*2 = 32768
  f16*   Qw  = (f16*)(ws + 163840);
  f16*   Kw  = (f16*)(ws + 163840 + QKVG);
  f16*   Vt  = (f16*)(ws + 163840 + 2 * QKVG);
  f16*   Gw  = (f16*)(ws + 163840 + 3 * QKVG);
  float* Bi  = (float*)(ws + 163840 + 4 * QKVG);   // 4*384*384*4 = 2,359,296

  const float qscale = 0.17677669529663687f;  // 1/sqrt(32)

  prep_weights<<<320, 256, 0, stream>>>(Wq, Wk, Wv, Wg, Wo, Wt, Wot, qscale);
  proj_kernel<<<(N_TOK * N_TOK) / 64, 64, 0, stream>>>(Zraw, gamma, beta, bw, bg,
                                                       Wt, Qw, Kw, Vt, Gw, Bi);
  attn_kernel<<<dim3(N_TOK / 16, N_TOK), 128, 0, stream>>>(Zraw, Qw, Kw, Vt, Gw,
                                                           Bi, Wot, ob, out);
}
